// DistanceSelfAttention_57801669869817
// MI455X (gfx1250) — matrix operand order repaired and checked by reading, not compiled, not run on hardware
//
#include <hip/hip_runtime.h>

// DistanceSelfAttention for gfx1250 (MI455X), bf16 WMMA + async-LDS path.
// B=16, N=512, E=512, H=8, D=64.
#define BB 16
#define NN 512
#define EE 512
#define HH 8
#define DD 64
#define NEGV -1.0e9f

typedef __attribute__((ext_vector_type(16))) __bf16 v16bf;
typedef __attribute__((ext_vector_type(8)))  float  v8f;

union Frag { v16bf v; unsigned u[8]; };

__device__ __forceinline__ v8f wmma_bf16(const Frag& a, const Frag& b, v8f c) {
  return __builtin_amdgcn_wmma_f32_16x16x32_bf16(false, a.v, false, b.v,
                                                 (short)0, c, false, false);
}

// round-to-nearest-even f32 -> bf16 (pair-packed / single)
__device__ __forceinline__ unsigned pk2(float lo, float hi) {
  unsigned a = __builtin_bit_cast(unsigned, lo);
  unsigned b = __builtin_bit_cast(unsigned, hi);
  a = (a + 0x7FFFu + ((a >> 16) & 1u)) >> 16;
  b = (b + 0x7FFFu + ((b >> 16) & 1u)) >> 16;
  return (b << 16) | (a & 0xFFFFu);
}
__device__ __forceinline__ unsigned short bf1(float x) {
  unsigned a = __builtin_bit_cast(unsigned, x);
  return (unsigned short)((a + 0x7FFFu + ((a >> 16) & 1u)) >> 16);
}

// gfx1250 16-bit A-matrix 16x32 layout: K index for dword p (0..7), lane l.
__device__ __forceinline__ int a_k(int p, int l) {
  return ((p & 4) << 2) + ((l >> 4) << 3) + ((p & 3) << 1);
}
// gfx1250 16-bit B-matrix 32x16 layout: same K order as A (lanes 0-15 K=0..7,16..23; lanes 16-31 K=8..15,24..31).
__device__ __forceinline__ int b_k(int p, int l) {
  return ((p & 4) << 2) + ((l >> 4) << 3) + ((p & 3) << 1);
}

// LDS byte offset of a generic pointer known to point into LDS.
typedef __attribute__((address_space(3))) const char lds_cchar;
__device__ __forceinline__ unsigned lds_addr(const void* p) {
  return (unsigned)(unsigned long long)(lds_cchar*)p;
}

__device__ __forceinline__ void loadA(Frag& A, const unsigned short* arow,
                                      int l, int kc) {
#pragma unroll
  for (int p = 0; p < 8; ++p)
    A.u[p] = *(const unsigned*)(arow + kc + a_k(p, l));
}
__device__ __forceinline__ void loadB(Frag& Bf, const unsigned short* brow,
                                      int l, int kc) {
#pragma unroll
  for (int p = 0; p < 8; ++p)
    Bf.u[p] = *(const unsigned*)(brow + kc + b_k(p, l));
}

// ---------------------------------------------------------------------------
// Pipelined 16x64-strip GEMM body: acc[dt] += A(row) * B(j0+dt*16..).
// Double-buffered so next k-chunk's b128 loads overlap current WMMAs
// (avoids the s_wait_loadcnt 0 drain before every wmma).
// arow = A base + row*EE ; brow[dt] = B base + (j0+dt*16+(l&15))*EE.
// ---------------------------------------------------------------------------
__device__ __forceinline__ void gemm_strip(const unsigned short* arow,
                                           const unsigned short* b0,
                                           const unsigned short* b1,
                                           const unsigned short* b2,
                                           const unsigned short* b3,
                                           int l, v8f acc[4]) {
  const unsigned short* brow[4] = {b0, b1, b2, b3};
  Frag A0, A1, B0[4], B1[4];
  loadA(A0, arow, l, 0);
#pragma unroll
  for (int dt = 0; dt < 4; ++dt) loadB(B0[dt], brow[dt], l, 0);
#pragma unroll
  for (int kc = 0; kc < EE; kc += 64) {
    // issue next chunk's loads before consuming current chunk
    loadA(A1, arow, l, kc + 32);
#pragma unroll
    for (int dt = 0; dt < 4; ++dt) loadB(B1[dt], brow[dt], l, kc + 32);
#pragma unroll
    for (int dt = 0; dt < 4; ++dt) acc[dt] = wmma_bf16(A0, B0[dt], acc[dt]);
    if (kc + 64 < EE) {
      loadA(A0, arow, l, kc + 64);
#pragma unroll
      for (int dt = 0; dt < 4; ++dt) loadB(B0[dt], brow[dt], l, kc + 64);
    }
#pragma unroll
    for (int dt = 0; dt < 4; ++dt) acc[dt] = wmma_bf16(A1, B1[dt], acc[dt]);
  }
}

// ---------------------------------------------------------------------------
// Pre-pass: fp32 -> bf16 (rne), 2 elements/thread.
// ---------------------------------------------------------------------------
__global__ void cvt_bf16(const float* __restrict__ src,
                         unsigned short* __restrict__ dst, int n2) {
  int i = blockIdx.x * blockDim.x + threadIdx.x;
  if (i < n2) {
    float2 v = *(const float2*)(src + 2 * (size_t)i);
    *(unsigned*)(dst + 2 * (size_t)i) = pk2(v.x, v.y);
  }
}

// ---------------------------------------------------------------------------
// Kernel A: Q/K/V projections (bf16 in, bf16 out). out = x @ w.T + b.
// 16x64 strip per wave. Q,K stored [B][H][N][D]; V stored [B][H][D][N].
// ---------------------------------------------------------------------------
__global__ void qkv_proj(const unsigned short* __restrict__ xb,
                         const unsigned short* __restrict__ wqb, const float* __restrict__ bq,
                         const unsigned short* __restrict__ wkb, const float* __restrict__ bk,
                         const unsigned short* __restrict__ wvb, const float* __restrict__ bv,
                         unsigned short* __restrict__ qo,
                         unsigned short* __restrict__ ko,
                         unsigned short* __restrict__ vt) {
  const int l = threadIdx.x & 31;
  const int wave = blockIdx.x * (blockDim.x >> 5) + (threadIdx.x >> 5);
  const int TI = (BB * NN) / 16;          // 512 row tiles
  const int SJ = EE / 64;                 // 8 col strips
  const int mat = wave / (TI * SJ);       // 0=Q 1=K 2=V
  const int rem = wave % (TI * SJ);
  const int i0 = (rem / SJ) * 16;
  const int j0 = (rem % SJ) * 64;
  const unsigned short* w = (mat == 0) ? wqb : (mat == 1) ? wkb : wvb;
  const float* bias       = (mat == 0) ? bq  : (mat == 1) ? bk  : bv;
  const unsigned short* arow = xb + (size_t)(i0 + (l & 15)) * EE;

  v8f acc[4] = {};
  gemm_strip(arow,
             w + (size_t)(j0 +  0 + (l & 15)) * EE,
             w + (size_t)(j0 + 16 + (l & 15)) * EE,
             w + (size_t)(j0 + 32 + (l & 15)) * EE,
             w + (size_t)(j0 + 48 + (l & 15)) * EE, l, acc);

  const int mhi = (l >> 4) << 3;
#pragma unroll
  for (int dt = 0; dt < 4; ++dt) {
    const int col = j0 + dt * 16 + (l & 15);
    const float bj = bias[col];
    const int h = col / DD, d = col % DD;
#pragma unroll
    for (int r = 0; r < 8; ++r) {
      const int i  = i0 + r + mhi;
      const int bb = i / NN, nr = i % NN;
      const unsigned short val = bf1(acc[dt][r] + bj);
      if (mat == 0)      qo[((size_t)(bb * HH + h) * NN + nr) * DD + d] = val;
      else if (mat == 1) ko[((size_t)(bb * HH + h) * NN + nr) * DD + d] = val;
      else               vt[((size_t)(bb * HH + h) * DD + d) * NN + nr] = val;
    }
  }
}

// ---------------------------------------------------------------------------
// Kernel B: fused attention. One (b,h,quarter) per 256-thread block.
// K strip (64KB) and V^T strip (64KB) are async-staged into LDS once and
// shared by 8 waves; each wave owns a 16-row query strip. Two-pass softmax
// on WMMA C-fragments (shfl-butterfly row reductions), normalization folded
// into the AV epilogue. LDS: 64K + 64K + 8*16K probs = 256KB of the 320KB WGP.
// ---------------------------------------------------------------------------
__global__ void __launch_bounds__(256) attn(
    const unsigned short* __restrict__ q,
    const unsigned short* __restrict__ k,
    const unsigned short* __restrict__ vt,
    const float* __restrict__ db,
    const int* __restrict__ mask,
    unsigned short* __restrict__ o) {
  extern __shared__ char smem[];
  unsigned short* kl = (unsigned short*)smem;                 // [N][D] bf16
  unsigned short* vl = (unsigned short*)(smem + 65536);       // [D][N] bf16
  const int l = threadIdx.x & 31;
  const int w = threadIdx.x >> 5;
  unsigned short* pw = (unsigned short*)(smem + 131072) + (size_t)w * 16 * NN;

  const int bh  = blockIdx.x >> 2;        // (b,h)
  const int qtr = blockIdx.x & 3;
  const int b = bh / HH;
  const int h = bh % HH;
  const int i0 = (qtr * 8 + w) * 16;
  const unsigned short* qh = q  + (size_t)bh * NN * DD;
  const unsigned short* kh = k  + (size_t)bh * NN * DD;
  const unsigned short* vh = vt + (size_t)bh * DD * NN;

  // ---- async stage K and V^T strips into LDS (ASYNCcnt path) ----
  {
    const unsigned kbase = lds_addr(kl);
    const unsigned vbase = lds_addr(vl);
    for (int ofs = threadIdx.x * 16; ofs < NN * DD * 2; ofs += 256 * 16) {
      unsigned long long ga = (unsigned long long)((const char*)kh + ofs);
      unsigned la = kbase + (unsigned)ofs;
      asm volatile("global_load_async_to_lds_b128 %0, %1, off"
                   :: "v"(la), "v"(ga) : "memory");
      unsigned long long gb = (unsigned long long)((const char*)vh + ofs);
      unsigned lb = vbase + (unsigned)ofs;
      asm volatile("global_load_async_to_lds_b128 %0, %1, off"
                   :: "v"(lb), "v"(gb) : "memory");
    }
    asm volatile("s_wait_asynccnt 0" ::: "memory");
    __syncthreads();
  }

  // Q strip A-fragments (K-dim = D = 64 -> two 32-chunks), kept resident.
  Frag qa[2];
#pragma unroll
  for (int kc2 = 0; kc2 < 2; ++kc2)
    loadA(qa[kc2], qh + (size_t)(i0 + (l & 15)) * DD, l, kc2 * 32);

  const int mhi = (l >> 4) << 3;
  const float scale = 0.125f;             // 1/sqrt(64)
  const float* dbb = db + (size_t)b * NN * NN;

  // ---- pass 1: row maxima of clip(QK^T*scale + db) with key mask ----
  float mrow[8];
#pragma unroll
  for (int r = 0; r < 8; ++r) mrow[r] = -3.0e38f;
  for (int jt = 0; jt < NN / 16; ++jt) {
    const int j = jt * 16 + (l & 15);
    v8f c = {};
#pragma unroll
    for (int kc2 = 0; kc2 < 2; ++kc2) {
      Frag kb;
      loadB(kb, kl + (size_t)(jt * 16 + (l & 15)) * DD, l, kc2 * 32);
      c = wmma_bf16(qa[kc2], kb, c);
    }
    const int mk = mask[b * NN + j];
#pragma unroll
    for (int r = 0; r < 8; ++r) {
      float s = c[r] * scale + dbb[(size_t)(i0 + r + mhi) * NN + j];
      s = fminf(10.f, fmaxf(-10.f, s));
      s = mk ? s : NEGV;
      mrow[r] = fmaxf(mrow[r], s);
    }
  }
#pragma unroll
  for (int r = 0; r < 8; ++r) {
    mrow[r] = fmaxf(mrow[r], __shfl_xor(mrow[r], 1, 32));
    mrow[r] = fmaxf(mrow[r], __shfl_xor(mrow[r], 2, 32));
    mrow[r] = fmaxf(mrow[r], __shfl_xor(mrow[r], 4, 32));
    mrow[r] = fmaxf(mrow[r], __shfl_xor(mrow[r], 8, 32));
  }

  // ---- pass 2: recompute scores, exp, accumulate row sums, store probs ----
  float sumr[8];
#pragma unroll
  for (int r = 0; r < 8; ++r) sumr[r] = 0.f;
  for (int jt = 0; jt < NN / 16; ++jt) {
    const int j = jt * 16 + (l & 15);
    v8f c = {};
#pragma unroll
    for (int kc2 = 0; kc2 < 2; ++kc2) {
      Frag kb;
      loadB(kb, kl + (size_t)(jt * 16 + (l & 15)) * DD, l, kc2 * 32);
      c = wmma_bf16(qa[kc2], kb, c);
    }
    const int mk = mask[b * NN + j];
#pragma unroll
    for (int r = 0; r < 8; ++r) {
      float s = c[r] * scale + dbb[(size_t)(i0 + r + mhi) * NN + j];
      s = fminf(10.f, fmaxf(-10.f, s));
      s = mk ? s : NEGV;
      float e = __expf(s - mrow[r]);
      sumr[r] += e;
      pw[(r + mhi) * NN + j] = bf1(e);    // unnormalized prob
    }
  }
  float oinv[8];
#pragma unroll
  for (int r = 0; r < 8; ++r) {
    float s = sumr[r];
    s += __shfl_xor(s, 1, 32);
    s += __shfl_xor(s, 2, 32);
    s += __shfl_xor(s, 4, 32);
    s += __shfl_xor(s, 8, 32);
    oinv[r] = 1.0f / s;
  }

  // ---- O = P @ V from LDS; normalization folded into epilogue ----
  v8f oacc[4] = {};
  for (int kc = 0; kc < NN; kc += 32) {
    Frag pa;
    loadA(pa, pw + (l & 15) * NN, l, kc);
#pragma unroll
    for (int dt = 0; dt < 4; ++dt) {
      Frag vb;
      loadB(vb, vl + (size_t)(dt * 16 + (l & 15)) * NN, l, kc);
      oacc[dt] = wmma_bf16(pa, vb, oacc[dt]);
    }
  }
#pragma unroll
  for (int dt = 0; dt < 4; ++dt)
#pragma unroll
    for (int r = 0; r < 8; ++r) {
      const int i = i0 + r + mhi;
      const int d = dt * 16 + (l & 15);
      o[(size_t)(b * NN + i) * EE + h * DD + d] = bf1(oacc[dt][r] * oinv[r]);
    }
}

// ---------------------------------------------------------------------------
// Kernel C: output projection. out = o @ wo.T + bo, fp32 result.
// 16x64 strip per wave, bf16 operands, pipelined GEMM body.
// ---------------------------------------------------------------------------
__global__ void out_proj(const unsigned short* __restrict__ o,
                         const unsigned short* __restrict__ wob,
                         const float* __restrict__ bo,
                         float* __restrict__ out) {
  const int l = threadIdx.x & 31;
  const int wave = blockIdx.x * (blockDim.x >> 5) + (threadIdx.x >> 5);
  const int SJ = EE / 64;
  const int i0 = (wave / SJ) * 16;
  const int j0 = (wave % SJ) * 64;
  const unsigned short* arow = o + (size_t)(i0 + (l & 15)) * EE;
  v8f acc[4] = {};
  gemm_strip(arow,
             wob + (size_t)(j0 +  0 + (l & 15)) * EE,
             wob + (size_t)(j0 + 16 + (l & 15)) * EE,
             wob + (size_t)(j0 + 32 + (l & 15)) * EE,
             wob + (size_t)(j0 + 48 + (l & 15)) * EE, l, acc);
  const int mhi = (l >> 4) << 3;
#pragma unroll
  for (int dt = 0; dt < 4; ++dt) {
    const int col = j0 + dt * 16 + (l & 15);
    const float bj = bo[col];
#pragma unroll
    for (int r = 0; r < 8; ++r)
      out[(size_t)(i0 + r + mhi) * EE + col] = acc[dt][r] + bj;
  }
}

// ---------------------------------------------------------------------------
extern "C" void kernel_launch(void* const* d_in, const int* in_sizes, int n_in,
                              void* d_out, int out_size, void* d_ws, size_t ws_size,
                              hipStream_t stream) {
  const float* x    = (const float*)d_in[0];
  const float* db   = (const float*)d_in[1];
  const int*   mask = (const int*)d_in[2];
  const float* wq   = (const float*)d_in[3];
  const float* bq   = (const float*)d_in[4];
  const float* wk   = (const float*)d_in[5];
  const float* bk   = (const float*)d_in[6];
  const float* wv   = (const float*)d_in[7];
  const float* bv   = (const float*)d_in[8];
  const float* wo   = (const float*)d_in[9];
  const float* bo   = (const float*)d_in[10];
  float* out = (float*)d_out;

  // workspace (bf16): xb 8MB | 4 weights 2MB | Q,K,V^T,O 32MB  -> ~42MB
  const size_t nx = (size_t)BB * NN * EE;       // 4M
  const size_t nw = (size_t)EE * EE;            // 256K
  unsigned short* us  = (unsigned short*)d_ws;
  unsigned short* xb  = us;
  unsigned short* wqb = xb + nx;
  unsigned short* wkb = wqb + nw;
  unsigned short* wvb = wkb + nw;
  unsigned short* wob = wvb + nw;
  unsigned short* qw  = wob + nw;
  unsigned short* kw  = qw + nx;
  unsigned short* vtw = kw + nx;
  unsigned short* ow  = vtw + nx;

  // fp32 -> bf16 pre-pass (x + 4 weight matrices)
  cvt_bf16<<<(int)((nx / 2 + 255) / 256), 256, 0, stream>>>(x, xb, (int)(nx / 2));
  cvt_bf16<<<(int)((nw / 2 + 255) / 256), 256, 0, stream>>>(wq, wqb, (int)(nw / 2));
  cvt_bf16<<<(int)((nw / 2 + 255) / 256), 256, 0, stream>>>(wk, wkb, (int)(nw / 2));
  cvt_bf16<<<(int)((nw / 2 + 255) / 256), 256, 0, stream>>>(wv, wvb, (int)(nw / 2));
  cvt_bf16<<<(int)((nw / 2 + 255) / 256), 256, 0, stream>>>(wo, wob, (int)(nw / 2));

  // QKV projections: 3 mats * 512 row-tiles * 8 col-strips = 12288 waves.
  qkv_proj<<<12288 / 4, 128, 0, stream>>>(xb, wqb, bq, wkb, bk, wvb, bv,
                                          qw, kw, vtw);
  // Attention: 128 (b,h) pairs x 4 quarters; 256 threads; 256KB dynamic LDS.
  attn<<<512, 256, 262144, stream>>>(qw, kw, vtw, db, mask, ow);
  // Output projection: 512 row-tiles * 8 col-strips = 4096 waves.
  out_proj<<<4096 / 4, 128, 0, stream>>>(ow, wob, bo, out);
}
